// Interact_10299331576088
// MI455X (gfx1250) — matrix operand order repaired and checked by reading, not compiled, not run on hardware
//
#include <hip/hip_runtime.h>
#include <hip/hip_bf16.h>

typedef __bf16 v16bf __attribute__((ext_vector_type(16)));
typedef __bf16 v8bf  __attribute__((ext_vector_type(8)));
typedef __bf16 v4bf  __attribute__((ext_vector_type(4)));
typedef __bf16 v2bf  __attribute__((ext_vector_type(2)));
typedef float  v8f   __attribute__((ext_vector_type(8)));
typedef unsigned int v4u __attribute__((ext_vector_type(4)));
typedef int    v8i   __attribute__((ext_vector_type(8)));
typedef int    v4i   __attribute__((ext_vector_type(4)));

static constexpr int kB  = 16;
static constexpr int kLQ = 2048;
static constexpr int kLV = 2048;
static constexpr int kD  = 1024;
#define NEG_BIG_F 1.0e10f

#define SHUF16(lo, hi) __builtin_shufflevector(lo, hi, 0,1,2,3,4,5,6,7,8,9,10,11,12,13,14,15)

// ---- CDNA5 TDM feature detection (probe-verified builtins, guarded) ----
#if __has_builtin(__builtin_amdgcn_tensor_load_to_lds) && __has_builtin(__builtin_amdgcn_s_wait_tensorcnt)
#define HAVE_TDM 1
#define WAIT_TENSORCNT(n) __builtin_amdgcn_s_wait_tensorcnt(n)
#else
#define HAVE_TDM 0
#define WAIT_TENSORCNT(n)
#endif

__device__ __forceinline__ unsigned lds_addr_of(const void* p) {
  // low 32 bits of a flat shared-space address are the LDS byte offset
  return (unsigned)(size_t)p;
}

// 2D bf16 tile DMA: global (row-major, dim0 contiguous) -> LDS, rows padded by
// 16B every 64B (pad_interval=16 dwords -> code 3, pad_amount=4 dwords -> code 3),
// i.e. a [tile1][32] tile lands as rows of stride 40 bf16.
__device__ __forceinline__ void tdm_load_2d_bf16(unsigned lds_byte_addr, const void* gsrc,
                                                 unsigned tensor_d0, unsigned tensor_d1,
                                                 unsigned tile_d0, unsigned tile_d1,
                                                 unsigned d0_stride) {
#if HAVE_TDM
  unsigned long long ga = (unsigned long long)(size_t)gsrc;
  v4u g0;
  g0[0] = 1u;                                            // count=1 (valid user D#)
  g0[1] = lds_byte_addr;                                 // lds_addr [63:32]
  g0[2] = (unsigned)(ga & 0xffffffffu);                  // global_addr [95:64]
  g0[3] = (unsigned)((ga >> 32) & 0x01ffffffu) | (2u << 30);  // addr[56:32] | type=2
  v8i g1;
  g1[0] = (int)((1u << 16) | (1u << 20) | (3u << 22) | (3u << 25)); // 2B | pad_en | int=16dw | amt=4dw
  g1[1] = (int)((tensor_d0 & 0xffffu) << 16);            // tensor_dim0 [63:48]
  g1[2] = (int)(((tensor_d0 >> 16) & 0xffffu) | ((tensor_d1 & 0xffffu) << 16));
  g1[3] = (int)(((tensor_d1 >> 16) & 0xffffu) | ((tile_d0 & 0xffffu) << 16));
  g1[4] = (int)(tile_d1 & 0xffffu);                      // tile_dim1, tile_dim2=0
  g1[5] = (int)d0_stride;                                // tensor_dim0_stride low32
  g1[6] = 0;
  g1[7] = 0;
  v4i z4 = {0, 0, 0, 0};
#if defined(__clang_major__) && __clang_major__ >= 23
  v8i z8 = {0, 0, 0, 0, 0, 0, 0, 0};
  __builtin_amdgcn_tensor_load_to_lds(g0, g1, z4, z4, z8, 0);
#else
  __builtin_amdgcn_tensor_load_to_lds(g0, g1, z4, z4, 0);
#endif
#else
  (void)lds_byte_addr; (void)gsrc; (void)tensor_d0; (void)tensor_d1;
  (void)tile_d0; (void)tile_d1; (void)d0_stride;
#endif
}

__device__ __forceinline__ v8f zero8() {
  v8f z;
  #pragma unroll
  for (int e = 0; e < 8; ++e) z[e] = 0.0f;
  return z;
}

__device__ __forceinline__ v8f wmma_bf16(v16bf a, v16bf b, v8f c) {
  return __builtin_amdgcn_wmma_f32_16x16x32_bf16(false, a, false, b, (short)0, c, false, false);
}

// order-preserving float -> uint map (exact, monotone); inverse in k_qmv
__device__ __forceinline__ unsigned fmapu(float f) {
  unsigned u = __float_as_uint(f);
  return (u & 0x80000000u) ? ~u : (u | 0x80000000u);
}

// ---------- prep kernels ----------

// kwT[n][k] = bf16( kernel[k][n] * 0.1 )  (32x32 LDS tile transpose)
__global__ void k_cvt_kernelT(const float* __restrict__ krn, __bf16* __restrict__ kwT) {
  __shared__ float t[32][33];
  int k0 = blockIdx.x * 32, n0 = blockIdx.y * 32;
  int c = threadIdx.x & 31, r0 = threadIdx.x >> 5;
  #pragma unroll
  for (int rr = 0; rr < 4; ++rr) {
    int r = r0 * 4 + rr;
    t[r][c] = krn[(size_t)(k0 + r) * kD + n0 + c];
  }
  __syncthreads();
  #pragma unroll
  for (int rr = 0; rr < 4; ++rr) {
    int r = r0 * 4 + rr;
    kwT[(size_t)(n0 + r) * kD + k0 + c] = (__bf16)(t[c][r] * 0.1f);
  }
}

__global__ void k_init_mv(unsigned* __restrict__ mvu) {
  int i = blockIdx.x * blockDim.x + threadIdx.x;
  if (i < kB * kD) mvu[i] = 0u;   // < fmapu(x) for every finite x
}

// vb = bf16(v), vbT = bf16(v)^T, and fused masked column-max into mvu (atomicMax)
__global__ void k_trans_v(const float* __restrict__ v, const float* __restrict__ vmask,
                          __bf16* __restrict__ vb, __bf16* __restrict__ vbT,
                          unsigned* __restrict__ mvu) {
  __shared__ float t[32][33];
  __shared__ float msk[32];
  int b = blockIdx.z;
  int key0 = blockIdx.x * 32, d0 = blockIdx.y * 32;
  int c = threadIdx.x & 31, r0 = threadIdx.x >> 5;
  if (threadIdx.x < 32)
    msk[threadIdx.x] = (1.0f - vmask[(size_t)b * kLV + key0 + threadIdx.x]) * NEG_BIG_F;
  #pragma unroll
  for (int rr = 0; rr < 4; ++rr) {
    int r = r0 * 4 + rr;
    float val = v[((size_t)b * kLV + key0 + r) * kD + d0 + c];
    t[r][c] = val;
    vb[((size_t)b * kLV + key0 + r) * kD + d0 + c] = (__bf16)val;
  }
  __syncthreads();
  #pragma unroll
  for (int rr = 0; rr < 4; ++rr) {
    int r = r0 * 4 + rr;
    vbT[((size_t)b * kD + d0 + r) * kLV + key0 + c] = (__bf16)t[c][r];
  }
  if (r0 == 0) {                 // 32 threads: masked max over this tile's 32 keys
    float m = -INFINITY;
    #pragma unroll
    for (int r = 0; r < 32; ++r)
      m = fmaxf(m, t[r][c] - msk[r]);
    atomicMax(&mvu[(size_t)b * kD + d0 + c], fmapu(m));
  }
}

// out[..., 1024:2048] = q ; out[..., 2048:3072] = unmap(mvu) broadcast
__global__ void k_qmv(const float* __restrict__ q, const unsigned* __restrict__ mvu,
                      float* __restrict__ out) {
  size_t i = (size_t)blockIdx.x * blockDim.x + threadIdx.x;
  if (i >= (size_t)kB * kLQ * kD) return;
  int d = (int)(i % kD);
  size_t row = i / kD;
  int b = (int)(row / kLQ);
  float* o = out + row * (size_t)(3 * kD);
  unsigned u = mvu[b * kD + d];
  unsigned bits = (u & 0x80000000u) ? (u ^ 0x80000000u) : ~u;
  o[kD + d]     = q[i];
  o[2 * kD + d] = __uint_as_float(bits);
}

// ---------- qw = bf16( q @ (0.1*kernel) ) ----------
// Block owns a full 64-row x 1024-col output slab: A panel staged once in LDS
// (q read exactly once), kwT streamed as a flat 256-tile TDM pipeline.
__global__ __launch_bounds__(256) void k_gemm_qw(const float* __restrict__ q,
                                                 const __bf16* __restrict__ kwT,
                                                 __bf16* __restrict__ qw) {
  __shared__ alignas(16) __bf16 Aall[64][kD + 8];   // 132 KB, 2064B rows (16B-aligned)
  __shared__ alignas(16) __bf16 Bs[2][128][40];     // 20 KB, 80B rows
  const int b  = blockIdx.y;
  const int m0 = blockIdx.x * 64;
  const int tid  = threadIdx.x;
  const int wave = tid >> 5, lane = tid & 31;
  const int half = lane >> 4, ln = lane & 15;
  const int mt = wave & 3;
  const int nb = (wave >> 2) * 64;
  const float* qb = q + (size_t)b * kLQ * kD;

  // stage the 64x1024 A panel once: float4 loads -> packed v4bf stores
  #pragma unroll 4
  for (int i = 0; i < 64; ++i) {
    int idx = i * 256 + tid;            // 16384 float4s
    int r = idx >> 8, c4 = idx & 255;
    float4 f = *(const float4*)&qb[(size_t)(m0 + r) * kD + c4 * 4];
    v4bf pk;
    pk[0] = (__bf16)f.x; pk[1] = (__bf16)f.y; pk[2] = (__bf16)f.z; pk[3] = (__bf16)f.w;
    *(v4bf*)&Aall[r][c4 * 4] = pk;
  }

  auto loadB = [&](int buf, int n0, int k0) {
#if HAVE_TDM
    if (wave == 0)
      tdm_load_2d_bf16(lds_addr_of(&Bs[buf][0][0]), kwT + (size_t)n0 * kD + k0,
                       kD, kD, 32, 128, kD);
#else
    #pragma unroll
    for (int i = 0; i < 16; ++i) {
      int idx = i * 256 + tid;
      int n = idx >> 5, c = idx & 31;
      Bs[buf][n][c] = kwT[(size_t)(n0 + n) * kD + k0 + c];
    }
#endif
  };

  v8f acc[4];
  #pragma unroll
  for (int i = 0; i < 4; ++i) acc[i] = zero8();

  constexpr int NT = 8 * 32;            // 8 n-slabs x 32 k-steps
  loadB(0, 0, 0);
  for (int t = 0; t < NT; ++t) {
    const int cur = t & 1;
    const int n0 = (t >> 5) * 128;
    const int k0 = (t & 31) * 32;
    if (t + 1 < NT) {
      loadB(1 - cur, ((t + 1) >> 5) * 128, ((t + 1) & 31) * 32);
      WAIT_TENSORCNT(1);                // oldest DMA (this step's tile) complete
    } else {
      WAIT_TENSORCNT(0);
    }
    __syncthreads();

    const v8bf* pa = (const v8bf*)&Aall[mt * 16 + ln][k0 + half * 16];
    v16bf a = SHUF16(pa[0], pa[1]);
    #pragma unroll
    for (int nt = 0; nt < 4; ++nt) {
      const v8bf* pb = (const v8bf*)&Bs[cur][nb + nt * 16 + ln][half * 16];
      v16bf bb = SHUF16(pb[0], pb[1]);
      acc[nt] = wmma_bf16(a, bb, acc[nt]);
    }
    __syncthreads();

    if ((t & 31) == 31) {               // slab done: write out, reset accumulators
      #pragma unroll
      for (int nt = 0; nt < 4; ++nt) {
        #pragma unroll
        for (int r = 0; r < 8; ++r) {
          int M = m0 + mt * 16 + half * 8 + r;
          int N = n0 + nb + nt * 16 + ln;
          qw[((size_t)b * kLQ + M) * kD + N] = (__bf16)acc[nt][r];
        }
        acc[nt] = zero8();
      }
    }
  }
}

// ---------- flash attention: out[..., 0:1024] ----------
// 256 threads (8 waves), 32 query rows/block, d split 128 per wave.
// V^T key-tile staged in LDS by the TDM, overlapped with score matmul + softmax.
__global__ __launch_bounds__(256) void k_attn(const __bf16* __restrict__ qw,
                                              const __bf16* __restrict__ vb,
                                              const __bf16* __restrict__ vbT,
                                              const float* __restrict__ vmask,
                                              float* __restrict__ out) {
  __shared__ alignas(16) __bf16 VsT[1024][40];   // [d][key], 80B rows
  __shared__ float  Ssc[32][33];
  __shared__ alignas(16) __bf16 Pb[32][32];
  __shared__ float  maskadd[32];
  __shared__ float  mrow[32], lrow[32], srow[32], inv[32];

  const int b  = blockIdx.y;
  const int q0 = blockIdx.x * 32;
  const int tid  = threadIdx.x;
  const int wave = tid >> 5, lane = tid & 31;
  const int half = lane >> 4, ln = lane & 15;
  const int dbase = wave * 128;
  const __bf16* qwb  = qw  + (size_t)b * kLQ * kD;
  const __bf16* vbb  = vb  + (size_t)b * kLV * kD;
  const __bf16* vbTb = vbT + (size_t)b * kD * kLV;

  // hoisted score A-fragments (loop-invariant over keys)
  v16bf afr[2][4];
  #pragma unroll
  for (int mt = 0; mt < 2; ++mt)
    #pragma unroll
    for (int fs = 0; fs < 4; ++fs) {
      int fb = dbase + fs * 32;
      const v8bf* pa = (const v8bf*)&qwb[(size_t)(q0 + mt * 16 + ln) * kD + fb + half * 16];
      afr[mt][fs] = SHUF16(pa[0], pa[1]);
    }

  v8f O[2][8];
  #pragma unroll
  for (int mt = 0; mt < 2; ++mt)
    #pragma unroll
    for (int nt = 0; nt < 8; ++nt) O[mt][nt] = zero8();
  if (tid < 32) { mrow[tid] = -INFINITY; lrow[tid] = 0.0f; }

  for (int key0 = 0; key0 < kLV; key0 += 32) {
    __syncthreads();                       // prior PV reads of VsT done
#if HAVE_TDM
    if (wave == 0)                          // [1024 d][32 key] tile -> padded LDS
      tdm_load_2d_bf16(lds_addr_of(&VsT[0][0]), vbTb + key0, kLV, kD, 32, 1024, kLV);
#else
    #pragma unroll
    for (int i = 0; i < 128; ++i) {
      int idx = i * 256 + tid;
      int dd = idx >> 5, c = idx & 31;
      VsT[dd][c] = vbTb[(size_t)dd * kLV + key0 + c];
    }
#endif
    #pragma unroll
    for (int i = 0; i < 4; ++i) {
      int idx = i * 256 + tid;
      Ssc[idx >> 5][idx & 31] = 0.0f;
    }
    if (tid < 32)
      maskadd[tid] = (1.0f - vmask[(size_t)b * kLV + key0 + tid]) * NEG_BIG_F;
    __syncthreads();

    // --- partial scores over this wave's 128-feature slice (B from global/L2) ---
    {
      if (key0 + 32 < kLV)
        __builtin_prefetch((const void*)&vbb[(size_t)(key0 + 32 + lane) * kD + dbase], 0, 0);
      v8f sacc[2][2];
      #pragma unroll
      for (int mt = 0; mt < 2; ++mt)
        #pragma unroll
        for (int nt = 0; nt < 2; ++nt) sacc[mt][nt] = zero8();
      for (int fs = 0; fs < 4; ++fs) {
        int fb = dbase + fs * 32;
        #pragma unroll
        for (int nt = 0; nt < 2; ++nt) {
          int key = key0 + nt * 16 + ln;
          const v8bf* pb = (const v8bf*)&vbb[(size_t)key * kD + fb + half * 16];
          v16bf bbf = SHUF16(pb[0], pb[1]);
          #pragma unroll
          for (int mt = 0; mt < 2; ++mt)
            sacc[mt][nt] = wmma_bf16(afr[mt][fs], bbf, sacc[mt][nt]);
        }
      }
      #pragma unroll
      for (int mt = 0; mt < 2; ++mt)
        #pragma unroll
        for (int nt = 0; nt < 2; ++nt)
          #pragma unroll
          for (int r = 0; r < 8; ++r)
            atomicAdd(&Ssc[mt * 16 + half * 8 + r][nt * 16 + ln], sacc[mt][nt][r]);
    }
    __syncthreads();

    // --- online softmax update (one thread per query row) ---
    if (tid < 32) {
      int row = tid;
      float mold = mrow[row];
      float mnew = mold;
      #pragma unroll
      for (int j = 0; j < 32; ++j)
        mnew = fmaxf(mnew, Ssc[row][j] - maskadd[j]);
      float sc = __expf(mold - mnew);
      float lsum = 0.0f;
      #pragma unroll
      for (int j = 0; j < 32; ++j) {
        float p = __expf(Ssc[row][j] - maskadd[j] - mnew);
        Pb[row][j] = (__bf16)p;
        lsum += p;
      }
      mrow[row] = mnew;
      lrow[row] = lrow[row] * sc + lsum;
      srow[row] = sc;
    }
    WAIT_TENSORCNT(0);                     // wave0's V-tile DMA complete
    __syncthreads();

    // --- rescale O and accumulate O += P @ V (V from LDS) ---
    v16bf a[2];
    #pragma unroll
    for (int mt = 0; mt < 2; ++mt) {
      const v8bf* pa = (const v8bf*)&Pb[mt * 16 + ln][half * 16];
      a[mt] = SHUF16(pa[0], pa[1]);
      #pragma unroll
      for (int r = 0; r < 8; ++r) {
        float sc = srow[mt * 16 + half * 8 + r];
        #pragma unroll
        for (int nt = 0; nt < 8; ++nt) O[mt][nt][r] *= sc;
      }
    }
    #pragma unroll
    for (int nt = 0; nt < 8; ++nt) {
      const v8bf* pb = (const v8bf*)&VsT[dbase + nt * 16 + ln][half * 16];
      v16bf bbf = SHUF16(pb[0], pb[1]);
      #pragma unroll
      for (int mt = 0; mt < 2; ++mt)
        O[mt][nt] = wmma_bf16(a[mt], bbf, O[mt][nt]);
    }
  }

  __syncthreads();
  if (tid < 32) inv[tid] = 1.0f / lrow[tid];
  __syncthreads();

  #pragma unroll
  for (int mt = 0; mt < 2; ++mt)
    #pragma unroll
    for (int r = 0; r < 8; ++r) {
      int M = q0 + mt * 16 + half * 8 + r;
      float s = inv[mt * 16 + half * 8 + r];
      size_t base = ((size_t)b * kLQ + M) * (size_t)(3 * kD);
      #pragma unroll
      for (int nt = 0; nt < 8; ++nt)
        out[base + dbase + nt * 16 + ln] = O[mt][nt][r] * s;
    }
}

// ---------- launch ----------
extern "C" void kernel_launch(void* const* d_in, const int* in_sizes, int n_in,
                              void* d_out, int out_size, void* d_ws, size_t ws_size,
                              hipStream_t stream) {
  (void)in_sizes; (void)n_in; (void)out_size; (void)ws_size;
  const float* q     = (const float*)d_in[0];
  const float* v     = (const float*)d_in[1];
  const float* vmask = (const float*)d_in[2];
  const float* krn   = (const float*)d_in[3];
  float* out = (float*)d_out;

  char* ws = (char*)d_ws;
  __bf16*   kwT = (__bf16*)  (ws);                        //   2 MiB : (0.1*kernel)^T
  __bf16*   vb  = (__bf16*)  (ws + ((size_t)2   << 20));  //  64 MiB : v, bf16
  __bf16*   qw  = (__bf16*)  (ws + ((size_t)66  << 20));  //  64 MiB : q@kernel*0.1
  unsigned* mvu = (unsigned*)(ws + ((size_t)130 << 20));  //  64 KiB : mapped col-max
  __bf16*   vbT = (__bf16*)  (ws + ((size_t)131 << 20));  //  64 MiB : v^T, bf16

  dim3 gk(kD / 32, kD / 32);
  k_cvt_kernelT<<<gk, 256, 0, stream>>>(krn, kwT);
  k_init_mv<<<(kB * kD) / 256, 256, 0, stream>>>(mvu);
  dim3 gt(kLV / 32, kD / 32, kB);
  k_trans_v<<<gt, 256, 0, stream>>>(v, vmask, vb, vbT, mvu);
  k_qmv<<<(kB * kLQ * kD) / 256, 256, 0, stream>>>(q, mvu, out);

  dim3 gq(kLQ / 64, kB);
  k_gemm_qw<<<gq, 256, 0, stream>>>(q, kwT, qw);

  dim3 ga(kLQ / 32, kB);
  k_attn<<<ga, 256, 0, stream>>>(qw, vb, vbT, vmask, out);
}
